// RobustMultiHeadAttention_21036749815889
// MI455X (gfx1250) — matrix operand order repaired and checked by reading, not compiled, not run on hardware
//
#include <hip/hip_runtime.h>
#include <math.h>

#define DM   1024
#define NH   16
#define HD   64
#define TT   2048
#define BB   4
#define NEGV (-10000.0f)

typedef _Float16 v16h __attribute__((ext_vector_type(16)));
typedef _Float16 v8h  __attribute__((ext_vector_type(8)));
typedef float    v8f  __attribute__((ext_vector_type(8)));
typedef float    v4f  __attribute__((ext_vector_type(4)));

#if defined(__has_builtin)
#if __has_builtin(__builtin_amdgcn_tensor_load_to_lds) && \
    __has_builtin(__builtin_amdgcn_s_wait_tensorcnt)
#define HAVE_TDM 1
#endif
#endif
#ifndef HAVE_TDM
#define HAVE_TDM 0
#endif

// ---- helpers -------------------------------------------------------------

// load 8 consecutive f32, convert to 8 f16
__device__ __forceinline__ v8h cvt8(const float* __restrict__ p) {
  v4f a = *reinterpret_cast<const v4f*>(p);
  v4f b = *reinterpret_cast<const v4f*>(p + 4);
  v8h r;
  r[0] = (_Float16)a[0]; r[1] = (_Float16)a[1];
  r[2] = (_Float16)a[2]; r[3] = (_Float16)a[3];
  r[4] = (_Float16)b[0]; r[5] = (_Float16)b[1];
  r[6] = (_Float16)b[2]; r[7] = (_Float16)b[3];
  return r;
}

__device__ __forceinline__ v16h cat16(v8h x, v8h y) {
  return __builtin_shufflevector(x, y, 0,1,2,3,4,5,6,7,8,9,10,11,12,13,14,15);
}

__device__ __forceinline__ v8f wmma_f16(v16h a, v16h b, v8f c) {
  return __builtin_amdgcn_wmma_f32_16x16x32_f16(false, a, false, b,
                                                (short)0, c, false, false);
}

#if HAVE_TDM
typedef unsigned int u32x4 __attribute__((ext_vector_type(4)));
typedef int          i32x4 __attribute__((ext_vector_type(4)));
typedef int          i32x8 __attribute__((ext_vector_type(8)));

// TDM: DMA a 32(keys) x 64(dims) f16 tile (row stride 64 elems) into LDS.
// D# packing per CDNA5 ISA ch.8 (group0: count/lds/global/type, group1: dims).
__device__ __forceinline__ void tdm_load_k32x64(_Float16* ldsdst,
                                                const _Float16* gsrc) {
  const unsigned long long ga = (unsigned long long)(size_t)gsrc;
  u32x4 g0;
  g0[0] = 1u;                                         // count=1 (valid), user mode
  g0[1] = (unsigned)(size_t)ldsdst;                   // lds_addr (bytes, low 32)
  g0[2] = (unsigned)(ga & 0xFFFFFFFFull);             // global_addr[31:0]
  g0[3] = (unsigned)((ga >> 32) & 0x01FFFFFFull)      // global_addr[56:32]
        | (2u << 30);                                 // type = 2 ("image")
  i32x8 g1;
  g1[0] = 1 << 16;                                    // data_size = 2 bytes
  g1[1] = (int)(64u << 16);                           // tensor_dim0 = 64 (lo16)
  g1[2] = (int)(((unsigned)TT & 0xFFFFu) << 16);      // tensor_dim1 = T (lo16)
  g1[3] = (int)(64u << 16);                           // tile_dim0 = 64
  g1[4] = 32;                                         // tile_dim1 = 32, tile_dim2=0
  g1[5] = 64;                                         // tensor_dim0_stride = 64
  g1[6] = 0;
  g1[7] = 0;
  i32x4 g2 = {0, 0, 0, 0};
  i32x4 g3 = {0, 0, 0, 0};
#if defined(__clang_major__) && (__clang_major__ >= 23)
  i32x8 g4 = {0, 0, 0, 0, 0, 0, 0, 0};
  __builtin_amdgcn_tensor_load_to_lds(g0, g1, g2, g3, g4, 0);
#else
  __builtin_amdgcn_tensor_load_to_lds(g0, g1, g2, g3, 0);
#endif
}
#endif

// ---- projection GEMM, 4x2 register-blocked: 64x32 C-tile per wave --------
// C[m,n] = sum_k A[m,k] * W[n,k]  (f32 in, f16 out)
// MODE 0: store head-major  out[b][h][t][d]   (d = n&63, h = n>>6)
// MODE 1: store transposed  out[b][h][d][t]   (for V)
template <int MODE>
__global__ __launch_bounds__(256) void proj_gemm(const float* __restrict__ A,
                                                 const float* __restrict__ W,
                                                 _Float16* __restrict__ out,
                                                 float scale) {
  const int wid  = blockIdx.x * (blockDim.x >> 5) + (threadIdx.x >> 5);
  const int lane = threadIdx.x & 31;
  const int hi = lane >> 4, lo = lane & 15;
  const int tilesN = DM / 32;                  // 32 N-blocks of width 32
  const int tm = wid / tilesN, tn = wid % tilesN;
  const int m0 = tm * 64, n0 = tn * 32;

  const float* arow[4];
  const float* brow[2];
  #pragma unroll
  for (int mi = 0; mi < 4; ++mi) arow[mi] = A + (size_t)(m0 + mi * 16 + lo) * DM;
  #pragma unroll
  for (int ni = 0; ni < 2; ++ni) brow[ni] = W + (size_t)(n0 + ni * 16 + lo) * DM;

  v8f acc[4][2];
  #pragma unroll
  for (int mi = 0; mi < 4; ++mi)
    #pragma unroll
    for (int ni = 0; ni < 2; ++ni) acc[mi][ni] = (v8f){};

  for (int kc = 0; kc < DM; kc += 32) {
    if (kc + 32 < DM) {                               // hide L2 latency
      __builtin_prefetch(arow[0] + kc + 32 + hi * 8, 0, 3);
      __builtin_prefetch(brow[0] + kc + 32 + hi * 16, 0, 3);
    }
    v16h af[4], bf[2];
    #pragma unroll
    for (int mi = 0; mi < 4; ++mi)                    // A 16x32: k = hi*8+j (+16)
      af[mi] = cat16(cvt8(arow[mi] + kc + hi * 8),
                     cvt8(arow[mi] + kc + 16 + hi * 8));
    #pragma unroll
    for (int ni = 0; ni < 2; ++ni)                    // B 32x16: k = hi*8+j (+16)
      bf[ni] = cat16(cvt8(brow[ni] + kc + hi * 8),
                     cvt8(brow[ni] + kc + 16 + hi * 8));
    #pragma unroll
    for (int mi = 0; mi < 4; ++mi)
      #pragma unroll
      for (int ni = 0; ni < 2; ++ni)
        acc[mi][ni] = wmma_f16(af[mi], bf[ni], acc[mi][ni]);
  }

  #pragma unroll
  for (int mi = 0; mi < 4; ++mi) {
    #pragma unroll
    for (int ni = 0; ni < 2; ++ni) {
      const int n = n0 + ni * 16 + lo, h = n >> 6, d = n & 63;
      if (MODE == 0) {
        #pragma unroll
        for (int r = 0; r < 8; ++r) {
          const int m = m0 + mi * 16 + r + 8 * hi;    // C tile: VGPR r -> M = r + 8*hi
          const int bidx = m >> 11, t = m & (TT - 1);
          out[((size_t)((bidx * NH + h) * TT + t)) * HD + d] =
              (_Float16)(acc[mi][ni][r] * scale);
        }
      } else {
        const int m = m0 + mi * 16 + 8 * hi;
        const int bidx = m >> 11, t = m & (TT - 1);
        v8h pk;
        #pragma unroll
        for (int r = 0; r < 8; ++r) pk[r] = (_Float16)(acc[mi][ni][r] * scale);
        // 8 consecutive t values -> one 16B store into Vt[b][h][d][t0..t0+7]
        *reinterpret_cast<v8h*>(out + ((size_t)((bidx * NH + h) * HD + d)) * TT + t) = pk;
      }
    }
  }
}

// ---- flash attention: one wave per (b, h, 16-query tile) -----------------
__global__ __launch_bounds__(256) void attn_kernel(const _Float16* __restrict__ Qh,
                                                   const _Float16* __restrict__ Kh,
                                                   const _Float16* __restrict__ Vt,
                                                   const float* __restrict__ amask,
                                                   const int* __restrict__ mfut,
                                                   _Float16* __restrict__ Ctx) {
  __shared__ _Float16 plds[8][16 * 32];          // per-wave P staging (C->A relayout)
#if HAVE_TDM
  __shared__ _Float16 kstage[8][32 * HD];        // per-wave TDM-staged K tile (4 KB)
#endif
  const int wslot = threadIdx.x >> 5;
  const int wid   = blockIdx.x * 8 + wslot;
  const int lane  = threadIdx.x & 31;
  const int hi = lane >> 4, lo = lane & 15;

  const int qt = wid & 127;                       // query tile (T/16 = 128)
  const int bh = wid >> 7;                        // b*NH + h
  const int bidx = bh >> 4, h = bh & 15;
  const int mf = mfut[0];

  // Q tile as two A-frags (d 0..31 and 32..63), reused every iteration
  const _Float16* qr = Qh + ((size_t)bh * TT + qt * 16 + lo) * HD;
  const v16h aq0 = cat16(*(const v8h*)(qr + hi * 16),      *(const v8h*)(qr + hi * 16 + 8));
  const v16h aq1 = cat16(*(const v8h*)(qr + 32 + hi * 16), *(const v8h*)(qr + 40 + hi * 16));

  float mrun[8], lrun[8];
  v8f o0 = {}, o1 = {}, o2 = {}, o3 = {};
  #pragma unroll
  for (int r = 0; r < 8; ++r) { mrun[r] = -INFINITY; lrun[r] = 0.0f; }

  const int nk = qt * 16 + 16;                    // #valid keys (causal upper bound)
  const int ntiles = mf ? ((nk + 31) >> 5) : (TT >> 5);
  const _Float16* kbase = Kh + (size_t)bh * TT * HD;
  const _Float16* vbase = Vt + (size_t)bh * HD * TT;
  const float*    mk    = amask + (size_t)bidx * TT;
  _Float16* P = &plds[wslot][0];
#if HAVE_TDM
  _Float16* KB = &kstage[wslot][0];
#endif

  for (int kt = 0; kt < ntiles; ++kt) {
    const int kb = kt * 32;
    // scores: S = Q(16x64) * K^T ; B-frag lane (hi,lo): key=lo(+16), d=hi*16+e
    v8f z = {};
#if HAVE_TDM
    // TDM: stream the 32x64 K tile into LDS (TENSORcnt), then consume via DS.
    tdm_load_k32x64(KB, kbase + (size_t)kb * HD);
    __builtin_amdgcn_s_wait_tensorcnt(0);
    const _Float16* kra = KB + (size_t)lo * HD;
    const _Float16* krb = KB + (size_t)(16 + lo) * HD;
#else
    const _Float16* kra = kbase + (size_t)(kb + lo) * HD;
    const _Float16* krb = kra + 16 * HD;
#endif
    v8f sa = wmma_f16(aq0, *(const v16h*)(kra + hi * 16), z);
    sa      = wmma_f16(aq1, *(const v16h*)(kra + 32 + hi * 16), sa);
    v8f sb = wmma_f16(aq0, *(const v16h*)(krb + hi * 16), z);
    sb      = wmma_f16(aq1, *(const v16h*)(krb + 32 + hi * 16), sb);

    const float pada = mk[kb + lo];
    const float padb = mk[kb + 16 + lo];
    const int ka = kb + lo, kbb = kb + 16 + lo;

    float rmax[8];
    #pragma unroll
    for (int r = 0; r < 8; ++r) {
      const int qm = qt * 16 + r + 8 * hi;
      float x = sa[r]; if (mf && ka  > qm) x += NEGV; if (pada == 0.0f) x = NEGV;
      float y = sb[r]; if (mf && kbb > qm) y += NEGV; if (padb == 0.0f) y = NEGV;
      sa[r] = x; sb[r] = y;
      float t = fmaxf(x, y);                      // row reduction: 16-lane butterfly
      t = fmaxf(t, __shfl_xor(t, 1, 16));
      t = fmaxf(t, __shfl_xor(t, 2, 16));
      t = fmaxf(t, __shfl_xor(t, 4, 16));
      t = fmaxf(t, __shfl_xor(t, 8, 16));
      rmax[r] = t;
    }

    float alpha[8], pa[8], pb[8];
    #pragma unroll
    for (int r = 0; r < 8; ++r) {
      const float nm = fmaxf(mrun[r], rmax[r]);
      alpha[r] = __expf(mrun[r] - nm);
      mrun[r]  = nm;
      pa[r] = __expf(sa[r] - nm);
      pb[r] = __expf(sb[r] - nm);
      float rs = pa[r] + pb[r];
      rs += __shfl_xor(rs, 1, 16);
      rs += __shfl_xor(rs, 2, 16);
      rs += __shfl_xor(rs, 4, 16);
      rs += __shfl_xor(rs, 8, 16);
      lrun[r] = lrun[r] * alpha[r] + rs;
    }
    #pragma unroll
    for (int r = 0; r < 8; ++r) {
      o0[r] *= alpha[r]; o1[r] *= alpha[r]; o2[r] *= alpha[r]; o3[r] *= alpha[r];
    }

    // probs: C-layout -> LDS -> A-layout
    #pragma unroll
    for (int r = 0; r < 8; ++r) {
      P[(r + 8 * hi) * 32 + lo]      = (_Float16)pa[r];
      P[(r + 8 * hi) * 32 + 16 + lo] = (_Float16)pb[r];
    }
    const v16h ap = cat16(*(const v8h*)&P[lo * 32 + hi * 16],
                          *(const v8h*)&P[lo * 32 + hi * 16 + 8]);

    // PV: B-frag lane (hi,lo): n = dt*16+lo (out dim), k = kb + hi*16 + e (contiguous in Vt)
    const _Float16* vr = vbase + (size_t)lo * TT + kb + hi * 16;
    o0 = wmma_f16(ap, *(const v16h*)(vr + (size_t)(0 * 16) * TT), o0);
    o1 = wmma_f16(ap, *(const v16h*)(vr + (size_t)(1 * 16) * TT), o1);
    o2 = wmma_f16(ap, *(const v16h*)(vr + (size_t)(2 * 16) * TT), o2);
    o3 = wmma_f16(ap, *(const v16h*)(vr + (size_t)(3 * 16) * TT), o3);
  }

  // normalize and store merged-head ctx [b][t][h*64+d] as f16
  float inv[8];
  #pragma unroll
  for (int r = 0; r < 8; ++r) inv[r] = 1.0f / lrun[r];
  #pragma unroll
  for (int r = 0; r < 8; ++r) {
    const int t = qt * 16 + r + 8 * hi;
    _Float16* dst = Ctx + ((size_t)(bidx * TT + t)) * DM + h * HD + lo;
    dst[0]  = (_Float16)(o0[r] * inv[r]);
    dst[16] = (_Float16)(o1[r] * inv[r]);
    dst[32] = (_Float16)(o2[r] * inv[r]);
    dst[48] = (_Float16)(o3[r] * inv[r]);
  }
}

// ---- output GEMM, 4x2 blocked: out[m,n] = sum_k Ctx[m,k] * Wo[n,k] -------
__global__ __launch_bounds__(256) void out_gemm(const _Float16* __restrict__ A,
                                                const float* __restrict__ W,
                                                float* __restrict__ out) {
  const int wid  = blockIdx.x * (blockDim.x >> 5) + (threadIdx.x >> 5);
  const int lane = threadIdx.x & 31;
  const int hi = lane >> 4, lo = lane & 15;
  const int tilesN = DM / 32;
  const int tm = wid / tilesN, tn = wid % tilesN;
  const int m0 = tm * 64, n0 = tn * 32;

  const _Float16* arow[4];
  const float*    brow[2];
  #pragma unroll
  for (int mi = 0; mi < 4; ++mi) arow[mi] = A + (size_t)(m0 + mi * 16 + lo) * DM;
  #pragma unroll
  for (int ni = 0; ni < 2; ++ni) brow[ni] = W + (size_t)(n0 + ni * 16 + lo) * DM;

  v8f acc[4][2];
  #pragma unroll
  for (int mi = 0; mi < 4; ++mi)
    #pragma unroll
    for (int ni = 0; ni < 2; ++ni) acc[mi][ni] = (v8f){};

  for (int kc = 0; kc < DM; kc += 32) {
    if (kc + 32 < DM) {
      __builtin_prefetch(arow[0] + kc + 32 + hi * 8, 0, 3);
      __builtin_prefetch(brow[0] + kc + 32 + hi * 16, 0, 3);
    }
    v16h af[4], bf[2];
    #pragma unroll
    for (int mi = 0; mi < 4; ++mi)
      af[mi] = cat16(*(const v8h*)(arow[mi] + kc + hi * 8),
                     *(const v8h*)(arow[mi] + kc + 16 + hi * 8));
    #pragma unroll
    for (int ni = 0; ni < 2; ++ni)
      bf[ni] = cat16(cvt8(brow[ni] + kc + hi * 8),
                     cvt8(brow[ni] + kc + 16 + hi * 8));
    #pragma unroll
    for (int mi = 0; mi < 4; ++mi)
      #pragma unroll
      for (int ni = 0; ni < 2; ++ni)
        acc[mi][ni] = wmma_f16(af[mi], bf[ni], acc[mi][ni]);
  }

  #pragma unroll
  for (int mi = 0; mi < 4; ++mi) {
    #pragma unroll
    for (int ni = 0; ni < 2; ++ni) {
      const int n = n0 + ni * 16 + lo;
      #pragma unroll
      for (int r = 0; r < 8; ++r) {
        const int m = m0 + mi * 16 + r + 8 * hi;
        out[(size_t)m * DM + n] = acc[mi][ni][r];
      }
    }
  }
}

// ---- launcher ------------------------------------------------------------
extern "C" void kernel_launch(void* const* d_in, const int* in_sizes, int n_in,
                              void* d_out, int out_size, void* d_ws, size_t ws_size,
                              hipStream_t stream) {
  const float* q  = (const float*)d_in[0];
  const float* k  = (const float*)d_in[1];
  const float* v  = (const float*)d_in[2];
  const float* am = (const float*)d_in[3];
  const float* Wq = (const float*)d_in[4];
  const float* Wk = (const float*)d_in[5];
  const float* Wv = (const float*)d_in[6];
  const float* Wo = (const float*)d_in[7];
  const int*   mf = (const int*)d_in[8];

  const size_t TEN = (size_t)BB * NH * TT * HD;   // 8,388,608 elems per tensor
  _Float16* Qh  = (_Float16*)d_ws;
  _Float16* Kh  = Qh + TEN;
  _Float16* Vt  = Kh + TEN;
  _Float16* Ctx = Vt + TEN;                        // [B][T][DM], same elem count

  const int gemm_waves = (BB * TT / 64) * (DM / 32);   // 128 * 32 = 4096
  const dim3 blk(256);
  const dim3 gemm_grid(gemm_waves / 8);                // 512 blocks, 8 waves each

  const float qscale = 0.125f;                         // (1/sqrt(sqrt(64)))^2
  proj_gemm<0><<<gemm_grid, blk, 0, stream>>>(q, Wq, Qh, qscale);
  proj_gemm<0><<<gemm_grid, blk, 0, stream>>>(k, Wk, Kh, 1.0f);
  proj_gemm<1><<<gemm_grid, blk, 0, stream>>>(v, Wv, Vt, 1.0f);

  const int attn_waves = BB * NH * (TT / 16);          // 8192
  attn_kernel<<<attn_waves / 8, blk, 0, stream>>>(Qh, Kh, Vt, am, mf, Ctx);

  out_gemm<<<gemm_grid, blk, 0, stream>>>(Ctx, Wo, (float*)d_out);
}
